// TinyCharGPT_33895881900264
// MI455X (gfx1250) — matrix operand order repaired and checked by reading, not compiled, not run on hardware
//
#include <hip/hip_runtime.h>

typedef __attribute__((ext_vector_type(16))) __bf16 v16bf;
typedef __attribute__((ext_vector_type(8)))  __bf16 v8bf;
typedef __attribute__((ext_vector_type(8)))  float  v8f;
typedef __attribute__((ext_vector_type(4)))  unsigned int u32x4;
typedef __attribute__((ext_vector_type(8)))  int i32x8;
typedef __attribute__((ext_vector_type(4)))  int i32x4;

union AF { v16bf v; v8bf h2[2]; };

// Load 8 consecutive f32 (global or LDS) and convert to packed bf16.
static __device__ __forceinline__ v8bf cvt8(const float* p) {
  float4 f0 = *(const float4*)(p);
  float4 f1 = *(const float4*)(p + 4);
  v8bf r;
  r[0] = (__bf16)f0.x; r[1] = (__bf16)f0.y; r[2] = (__bf16)f0.z; r[3] = (__bf16)f0.w;
  r[4] = (__bf16)f1.x; r[5] = (__bf16)f1.y; r[6] = (__bf16)f1.z; r[7] = (__bf16)f1.w;
  return r;
}

// --------------- Tensor Data Mover: 2D f32 tile (32 x 128 rows) -> LDS ---------------
// D# packing per CDNA5 ISA.
static __device__ __forceinline__ void tdm_load_a(
    unsigned lds_addr, const float* g, int row_stride_elems, int rem_rows, int rem_cols)
{
  const unsigned long long ga = (unsigned long long)(uintptr_t)g;
  u32x4 g0;
  g0[0] = 1u;                                                 // count=1 (valid user D#)
  g0[1] = lds_addr;                                           // lds_addr (bytes)
  g0[2] = (unsigned)ga;                                       // global_addr[31:0]
  g0[3] = (unsigned)((ga >> 32) & 0x01FFFFFFu) | (2u << 30);  // global_addr[56:32] | type=2
  i32x8 g1;
  const unsigned d0 = (unsigned)rem_cols;                     // tensor_dim0 (elements)
  const unsigned d1 = (unsigned)rem_rows;                     // tensor_dim1 (rows)
  g1[0] = (int)(2u << 16);                                    // data_size=4B, wg_mask=0
  g1[1] = (int)(d0 << 16);                                    // tensor_dim0[15:0] @ [63:48]
  g1[2] = (int)((d0 >> 16) | (d1 << 16));                     // dim0[31:16] | dim1[15:0]
  g1[3] = (int)((d1 >> 16) | (32u << 16));                    // dim1[31:16] | tile_dim0=32
  g1[4] = (int)128u;                                          // tile_dim1=128, tile_dim2=0
  g1[5] = (int)(unsigned)row_stride_elems;                    // tensor_dim0_stride[31:0]
  g1[6] = 0; g1[7] = 0;
  const i32x4 z4 = {0, 0, 0, 0};
#if __clang_major__ >= 23
  const i32x8 z8 = {0, 0, 0, 0, 0, 0, 0, 0};
  __builtin_amdgcn_tensor_load_to_lds(g0, g1, z4, z4, z8, 0);
#else
  __builtin_amdgcn_tensor_load_to_lds(g0, g1, z4, z4, 0);
#endif
}

// ------------------------- embedding: x = tok_emb[idx] + pos_emb -------------------------
__global__ __launch_bounds__(256) void embed_kernel(const int* __restrict__ idx,
    const float* __restrict__ tok, const float* __restrict__ pos, float* __restrict__ x)
{
  int i  = blockIdx.x * 256 + threadIdx.x;   // over B*T*E
  int e  = i % 768;
  int bt = i / 768;
  int t  = bt % 1024;
  x[i] = tok[idx[bt] * 768 + e] + pos[t * 768 + e];
}

// ------------------------- layernorm over E=768 -------------------------
__global__ __launch_bounds__(256) void ln_kernel(const float* __restrict__ x,
    const float* __restrict__ w, const float* __restrict__ b, float* __restrict__ out)
{
  __shared__ float rs[256], rq[256];
  const int row = blockIdx.x, tid = threadIdx.x;
  const float* xr = x + row * 768;
  float s = 0.f, s2 = 0.f;
  #pragma unroll
  for (int i = tid; i < 768; i += 256) { float v = xr[i]; s += v; s2 += v * v; }
  rs[tid] = s; rq[tid] = s2;
  __syncthreads();
  for (int o = 128; o > 0; o >>= 1) {
    if (tid < o) { rs[tid] += rs[tid + o]; rq[tid] += rq[tid + o]; }
    __syncthreads();
  }
  const float mu  = rs[0] * (1.f / 768.f);
  const float var = rq[0] * (1.f / 768.f) - mu * mu;
  const float inv = rsqrtf(var + 1e-5f);
  for (int i = tid; i < 768; i += 256)
    out[row * 768 + i] = (xr[i] - mu) * inv * w[i] + b[i];
}

// ------------------------- WMMA GEMM: C = act(A[M,K] @ W[K,N] + bias) (+res) -------------
// Block: 256 thr = 8 waves, 128x64 tile; wave (wm,wn) in 4x2 owns a 32x32 strip (4 WMMA/step).
// A tile (128x32 f32) is DMA'd into LDS by the Tensor Data Mover; W tile staged transposed bf16.
template<bool GELU, bool RES>
__global__ __launch_bounds__(256) void gemm_wmma_kernel(
    const float* __restrict__ A, const float* __restrict__ W,
    const float* __restrict__ bias, const float* __restrict__ res,
    float* __restrict__ C, int M, int N, int K)
{
  __shared__ float  As32[128 * 32];   // [m][k] row-major f32, filled by TDM
  __shared__ __bf16 Bs[64 * 32];      // transposed: [n][k] bf16
  const int tid  = threadIdx.x;
  const int wave = tid >> 5, lane = tid & 31;
  const int l16  = lane & 15, lh = lane >> 4;
  const int wm   = wave >> 1, wn = wave & 1;
  const int m0   = blockIdx.y * 128, n0 = blockIdx.x * 64;

  const v8f vz = {0.f,0.f,0.f,0.f,0.f,0.f,0.f,0.f};
  v8f acc[2][2] = {{vz, vz}, {vz, vz}};

  const int bk = tid >> 3;           // 0..31 (W k row)
  const int bn = (tid & 7) * 8;      // 0..56 (W n chunk)
  const unsigned ldsA = (unsigned)(uintptr_t)&As32[0];  // LDS byte offset (low 32 bits)

  for (int k0 = 0; k0 < K; k0 += 32) {
    // async DMA of the A tile into LDS (one TDM op per block per step)
    if (wave == 0)
      tdm_load_a(ldsA, A + (size_t)m0 * K + k0, K, M - m0, K - k0);
    // stage W tile transposed (coalesced read along n, scatter into [n][k])
    {
      const float* src = W + (k0 + bk) * N + n0 + bn;
      float4 f0 = *(const float4*)src;
      float4 f1 = *(const float4*)(src + 4);
      Bs[(bn + 0) * 32 + bk] = (__bf16)f0.x;
      Bs[(bn + 1) * 32 + bk] = (__bf16)f0.y;
      Bs[(bn + 2) * 32 + bk] = (__bf16)f0.z;
      Bs[(bn + 3) * 32 + bk] = (__bf16)f0.w;
      Bs[(bn + 4) * 32 + bk] = (__bf16)f1.x;
      Bs[(bn + 5) * 32 + bk] = (__bf16)f1.y;
      Bs[(bn + 6) * 32 + bk] = (__bf16)f1.z;
      Bs[(bn + 7) * 32 + bk] = (__bf16)f1.w;
      if (k0 + 32 < K) __builtin_prefetch(src + 32 * N, 0, 1);  // global_prefetch_b8
    }
    if (wave == 0) __builtin_amdgcn_s_wait_tensorcnt(0);
    __syncthreads();

    // A fragments (f32 LDS -> bf16): lane half 0 -> K{0..7,16..23}, half 1 -> K{8..15,24..31}
    AF a0, a1, b0, b1;
    const float* Af = As32 + (wm * 32 + l16) * 32 + (lh ? 8 : 0);
    a0.h2[0] = cvt8(Af);
    a0.h2[1] = cvt8(Af + 16);
    a1.h2[0] = cvt8(Af + 16 * 32);
    a1.h2[1] = cvt8(Af + 16 * 32 + 16);
    // B fragments: lane col = l16, half 0 -> K{0..7,16..23}, half 1 -> K{8..15,24..31}
    const int bc = (wn * 32 + l16) * 32 + (lh ? 8 : 0);
    b0.h2[0] = *(const v8bf*)&Bs[bc];
    b0.h2[1] = *(const v8bf*)&Bs[bc + 16];
    b1.h2[0] = *(const v8bf*)&Bs[bc + 16 * 32];
    b1.h2[1] = *(const v8bf*)&Bs[bc + 16 * 32 + 16];

    acc[0][0] = __builtin_amdgcn_wmma_f32_16x16x32_bf16(false, a0.v, false, b0.v, (short)0, acc[0][0], false, false);
    acc[0][1] = __builtin_amdgcn_wmma_f32_16x16x32_bf16(false, a0.v, false, b1.v, (short)0, acc[0][1], false, false);
    acc[1][0] = __builtin_amdgcn_wmma_f32_16x16x32_bf16(false, a1.v, false, b0.v, (short)0, acc[1][0], false, false);
    acc[1][1] = __builtin_amdgcn_wmma_f32_16x16x32_bf16(false, a1.v, false, b1.v, (short)0, acc[1][1], false, false);
    __syncthreads();
  }

  // epilogue per C layout: VGPR r -> row r (lanes 0-15) / r+8 (lanes 16-31), col = l16
  #pragma unroll
  for (int i = 0; i < 2; ++i) {
    const int crow = m0 + wm * 32 + i * 16 + lh * 8;
    #pragma unroll
    for (int j = 0; j < 2; ++j) {
      const int col = n0 + wn * 32 + j * 16 + l16;
      const float bi = bias ? bias[col] : 0.f;
      #pragma unroll
      for (int r = 0; r < 8; ++r) {
        float vv = acc[i][j][r] + bi;
        if (GELU) vv = 0.5f * vv * (1.f + erff(vv * 0.70710678118f));
        if (RES)  vv += res[(crow + r) * N + col];
        C[(crow + r) * N + col] = vv;
      }
    }
  }
}

// ------------------------- flash attention (causal), HS=64, T=1024 -------------------------
// Block: 128 thr = 4 waves; wave owns 16 query rows; streams key blocks of 32.
__global__ __launch_bounds__(128) void attn_kernel(
    const float* __restrict__ q, const float* __restrict__ k,
    const float* __restrict__ v, float* __restrict__ y)
{
  __shared__ __bf16 Vt[64 * 32];      // V transposed: [dim][key]
  __shared__ __bf16 Pb[4][16 * 32];   // per-wave P tile: [qrow][key]
  const v8f vz = {0.f,0.f,0.f,0.f,0.f,0.f,0.f,0.f};

  const int bidx = blockIdx.z, hd = blockIdx.y;
  const int tid = threadIdx.x, wave = tid >> 5, lane = tid & 31;
  const int l16 = lane & 15, lh = lane >> 4;
  const int q0  = blockIdx.x * 64 + wave * 16;

  // Q A-fragments (held in regs for the whole kernel)
  const float* qrowp = q + (size_t)(bidx * 1024 + q0 + l16) * 768 + hd * 64;
  AF qa0, qa1;
  {
    const int c0 = lh ? 8 : 0;
    qa0.h2[0] = cvt8(qrowp + c0);
    qa0.h2[1] = cvt8(qrowp + c0 + 16);
    qa1.h2[0] = cvt8(qrowp + 32 + c0);
    qa1.h2[1] = cvt8(qrowp + 32 + c0 + 16);
  }

  v8f o[4] = {vz, vz, vz, vz};
  float mrow[8], lrow[8];
  #pragma unroll
  for (int r = 0; r < 8; ++r) { mrow[r] = -3.0e38f; lrow[r] = 0.f; }

  const int numkb = blockIdx.x * 2 + 2;
  for (int kb = 0; kb < numkb; ++kb) {
    const int s0 = kb * 32;
    // stage V^T (all threads): 32 keys x 64 dims
    {
      const int key = tid >> 2;
      const int d0  = (tid & 3) * 16;
      const float* src = v + (size_t)(bidx * 1024 + s0 + key) * 768 + hd * 64 + d0;
      #pragma unroll
      for (int i = 0; i < 16; i += 4) {
        float4 f = *(const float4*)(src + i);
        Vt[(d0 + i + 0) * 32 + key] = (__bf16)f.x;
        Vt[(d0 + i + 1) * 32 + key] = (__bf16)f.y;
        Vt[(d0 + i + 2) * 32 + key] = (__bf16)f.z;
        Vt[(d0 + i + 3) * 32 + key] = (__bf16)f.w;
      }
    }
    __syncthreads();

    if (s0 <= q0 + 15) {  // causal: this wave still has live keys in this block
      // S = Q K^T : two 16-key subtiles, HS=64 -> 2 wmma each
      v8f sA = vz, sB = vz;
      {
        const float* kp0 = k + (size_t)(bidx * 1024 + s0 + l16) * 768 + hd * 64 + lh * 8;
        AF kf0, kf1;
        kf0.h2[0] = cvt8(kp0);      kf0.h2[1] = cvt8(kp0 + 16);
        kf1.h2[0] = cvt8(kp0 + 32); kf1.h2[1] = cvt8(kp0 + 48);
        sA = __builtin_amdgcn_wmma_f32_16x16x32_bf16(false, qa0.v, false, kf0.v, (short)0, sA, false, false);
        sA = __builtin_amdgcn_wmma_f32_16x16x32_bf16(false, qa1.v, false, kf1.v, (short)0, sA, false, false);
        const float* kp1 = kp0 + 16 * 768;
        kf0.h2[0] = cvt8(kp1);      kf0.h2[1] = cvt8(kp1 + 16);
        kf1.h2[0] = cvt8(kp1 + 32); kf1.h2[1] = cvt8(kp1 + 48);
        sB = __builtin_amdgcn_wmma_f32_16x16x32_bf16(false, qa0.v, false, kf0.v, (short)0, sB, false, false);
        sB = __builtin_amdgcn_wmma_f32_16x16x32_bf16(false, qa1.v, false, kf1.v, (short)0, sB, false, false);
      }

      // online softmax: row stats via xor-shuffles inside each 16-lane C group
      float p0v[8], p1v[8], alpha[8];
      #pragma unroll
      for (int r = 0; r < 8; ++r) {
        const int qq = q0 + r + lh * 8;
        float s0e = (s0 + l16      <= qq) ? sA[r] * 0.125f : -3.0e38f;
        float s1e = (s0 + 16 + l16 <= qq) ? sB[r] * 0.125f : -3.0e38f;
        float tm = fmaxf(s0e, s1e);
        tm = fmaxf(tm, __shfl_xor(tm, 1, 32));
        tm = fmaxf(tm, __shfl_xor(tm, 2, 32));
        tm = fmaxf(tm, __shfl_xor(tm, 4, 32));
        tm = fmaxf(tm, __shfl_xor(tm, 8, 32));
        const float nm = fmaxf(mrow[r], tm);
        alpha[r] = __expf(mrow[r] - nm);
        mrow[r] = nm;
        const float p0 = __expf(s0e - nm);
        const float p1 = __expf(s1e - nm);
        float ts = p0 + p1;
        ts += __shfl_xor(ts, 1, 32);
        ts += __shfl_xor(ts, 2, 32);
        ts += __shfl_xor(ts, 4, 32);
        ts += __shfl_xor(ts, 8, 32);
        lrow[r] = lrow[r] * alpha[r] + ts;
        p0v[r] = p0; p1v[r] = p1;
      }
      #pragma unroll
      for (int t = 0; t < 4; ++t) {
        #pragma unroll
        for (int r = 0; r < 8; ++r) o[t][r] *= alpha[r];
      }

      // bounce P (C layout) through LDS to rebuild it as an A fragment (16x32 keys)
      __bf16* P = Pb[wave];
      #pragma unroll
      for (int r = 0; r < 8; ++r) {
        P[(r + lh * 8) * 32 + l16]      = (__bf16)p0v[r];
        P[(r + lh * 8) * 32 + l16 + 16] = (__bf16)p1v[r];
      }
      asm volatile("s_wait_dscnt 0" ::: "memory");  // same-wave LDS RAW fence
      AF pa;
      const int pc = lh ? 8 : 0;
      pa.h2[0] = *(const v8bf*)&P[l16 * 32 + pc];
      pa.h2[1] = *(const v8bf*)&P[l16 * 32 + pc + 16];
      #pragma unroll
      for (int t = 0; t < 4; ++t) {
        AF vf;
        vf.h2[0] = *(const v8bf*)&Vt[(t * 16 + l16) * 32 + lh * 8];
        vf.h2[1] = *(const v8bf*)&Vt[(t * 16 + l16) * 32 + lh * 8 + 16];
        o[t] = __builtin_amdgcn_wmma_f32_16x16x32_bf16(false, pa.v, false, vf.v, (short)0, o[t], false, false);
      }
    }
    __syncthreads();  // protect Vt before next block's staging
  }

  // normalize and write y
  #pragma unroll
  for (int r = 0; r < 8; ++r) {
    const float inv = 1.f / lrow[r];
    float* dst = y + (size_t)(bidx * 1024 + q0 + r + lh * 8) * 768 + hd * 64;
    #pragma unroll
    for (int t = 0; t < 4; ++t) dst[t * 16 + l16] = o[t][r] * inv;
  }
}

// ------------------------- host orchestration -------------------------
extern "C" void kernel_launch(void* const* d_in, const int* in_sizes, int n_in,
                              void* d_out, int out_size, void* d_ws, size_t ws_size,
                              hipStream_t stream) {
  (void)in_sizes; (void)n_in; (void)out_size; (void)ws_size;
  const int   E = 768, T = 1024, Bn = 4, Ln = 6, E4 = 3072, Vv = 256;
  const int   M = Bn * T;                 // 4096 rows
  const size_t EE = (size_t)E * E;

  const int*   idx   = (const int*)d_in[0];
  const float* tok   = (const float*)d_in[1];
  const float* pos   = (const float*)d_in[2];
  const float* ln1w  = (const float*)d_in[3];
  const float* ln1b  = (const float*)d_in[4];
  const float* Wq    = (const float*)d_in[5];
  const float* bq    = (const float*)d_in[6];
  const float* Wk    = (const float*)d_in[7];
  const float* bk    = (const float*)d_in[8];
  const float* Wv    = (const float*)d_in[9];
  const float* bv    = (const float*)d_in[10];
  const float* Wp    = (const float*)d_in[11];
  const float* bp    = (const float*)d_in[12];
  const float* ln2w  = (const float*)d_in[13];
  const float* ln2b  = (const float*)d_in[14];
  const float* W1    = (const float*)d_in[15];
  const float* b1    = (const float*)d_in[16];
  const float* W2    = (const float*)d_in[17];
  const float* b2    = (const float*)d_in[18];
  const float* lnfw  = (const float*)d_in[19];
  const float* lnfb  = (const float*)d_in[20];
  const float* lmw   = (const float*)d_in[21];
  float* logits = (float*)d_out;

  float* xs = (float*)d_ws;
  float* hb = xs + (size_t)M * E;
  float* qb = hb + (size_t)M * E;
  float* kb = qb + (size_t)M * E;
  float* vb = kb + (size_t)M * E;
  float* mm = vb + (size_t)M * E;         // [M, 4E]

  embed_kernel<<<(M * E) / 256, 256, 0, stream>>>(idx, tok, pos, xs);

  const dim3 gE(E / 64, M / 128);         // 12 x 32
  const dim3 g4(E4 / 64, M / 128);        // 48 x 32
  const dim3 gV(Vv / 64, M / 128);        // 4 x 32
  const dim3 gA(T / 64, 12, Bn);          // attention: 16 x 12 x 4

  for (int l = 0; l < Ln; ++l) {
    ln_kernel<<<M, 256, 0, stream>>>(xs, ln1w + l * E, ln1b + l * E, hb);
    gemm_wmma_kernel<false, false><<<gE, 256, 0, stream>>>(hb, Wq + l * EE, bq + l * E, nullptr, qb, M, E, E);
    gemm_wmma_kernel<false, false><<<gE, 256, 0, stream>>>(hb, Wk + l * EE, bk + l * E, nullptr, kb, M, E, E);
    gemm_wmma_kernel<false, false><<<gE, 256, 0, stream>>>(hb, Wv + l * EE, bv + l * E, nullptr, vb, M, E, E);
    attn_kernel<<<gA, 128, 0, stream>>>(qb, kb, vb, hb);
    gemm_wmma_kernel<false, true ><<<gE, 256, 0, stream>>>(hb, Wp + l * EE, bp + l * E, xs, xs, M, E, E);
    ln_kernel<<<M, 256, 0, stream>>>(xs, ln2w + l * E, ln2b + l * E, hb);
    gemm_wmma_kernel<true,  false><<<g4, 256, 0, stream>>>(hb, W1 + (size_t)l * E * E4, b1 + l * E4, nullptr, mm, M, E4, E);
    gemm_wmma_kernel<false, true ><<<gE, 256, 0, stream>>>(mm, W2 + (size_t)l * E4 * E, b2 + l * E, xs, xs, M, E, E4);
  }
  ln_kernel<<<M, 256, 0, stream>>>(xs, lnfw, lnfb, hb);
  gemm_wmma_kernel<false, false><<<gV, 256, 0, stream>>>(hb, lmw, nullptr, nullptr, logits, M, Vv, E);
}
